// HybridAttention_22746146799820
// MI455X (gfx1250) — matrix operand order repaired and checked by reading, not compiled, not run on hardware
//
#include <hip/hip_runtime.h>
#include <hip/hip_bf16.h>

#define L_SEQ   4096
#define D_MODEL 1024
#define N_HEADS 16
#define N_AR    8
#define D_HEAD  64

typedef __attribute__((ext_vector_type(16))) __bf16 v16bf;
typedef __attribute__((ext_vector_type(8)))  float  v8f;
typedef __attribute__((ext_vector_type(4)))  __bf16 v4bf;
typedef __attribute__((ext_vector_type(2)))  __bf16 v2bf;

union FragBF { uint4 q[2]; v16bf v; };
union U2BF   { uint2 q; __bf16 h[4]; };

__device__ __forceinline__ v8f wmma_bf16(v16bf a, v16bf b, v8f c) {
  // D = A(16x32 bf16) * B(32x16 bf16) + C(16x16 f32)
  return __builtin_amdgcn_wmma_f32_16x16x32_bf16(false, a, false, b, (short)0, c, false, false);
}

__device__ __forceinline__ float red_max16(float x) {
  x = fmaxf(x, __shfl_xor(x, 1, 32));
  x = fmaxf(x, __shfl_xor(x, 2, 32));
  x = fmaxf(x, __shfl_xor(x, 4, 32));
  x = fmaxf(x, __shfl_xor(x, 8, 32));
  return x;
}
__device__ __forceinline__ float red_sum16(float x) {
  x += __shfl_xor(x, 1, 32);
  x += __shfl_xor(x, 2, 32);
  x += __shfl_xor(x, 4, 32);
  x += __shfl_xor(x, 8, 32);
  return x;
}

// Async global->LDS copy of 16 bytes (gfx1250 GLOBAL_LOAD_ASYNC_TO_LDS_B128,
// GV addressing; tracked by ASYNCcnt). ldsoff = LDS byte offset (generic
// pointer low 32 bits per the aperture mapping), ga = 64-bit global address.
__device__ __forceinline__ void async_copy_b128(unsigned ldsoff, const void* gp) {
  unsigned long long ga = (unsigned long long)gp;
  asm volatile("global_load_async_to_lds_b128 %0, %1, off"
               :: "v"(ldsoff), "v"(ga) : "memory");
}
__device__ __forceinline__ void wait_async0() {
  asm volatile("s_wait_asynccnt 0" ::: "memory");
}

__device__ __forceinline__ v2bf pack2(float a, float b) {
  v2bf p; p.x = (__bf16)a; p.y = (__bf16)b; return p;
}

// ---------------------------------------------------------------------------
// Kernel 1: Y = X @ W (f32 -> bf16 WMMA, f32 acc), optional RoPE,
// output stored bf16 as [head][seq][64].
// Block: 256 threads (8 waves). Block tile: 128 rows x 64 cols (one head).
// K staged 64 per barrier: 8 WMMAs per wave per stage.
// ---------------------------------------------------------------------------
__global__ __launch_bounds__(256) void gemm_qkv_rope(
    const float* __restrict__ X, const float* __restrict__ W,
    __bf16* __restrict__ Out, int apply_rope) {
  __shared__ __bf16 As[128][72];  // rows x 64k, padded (144B row, 16B aligned)
  __shared__ __bf16 Bt[64][72];   // [n][k] transposed
  const int tid  = threadIdx.x;
  const int wave = tid >> 5, lane = tid & 31, lo = lane & 15, hi = lane >> 4;
  const int n0 = blockIdx.x * 64;
  const int m0 = blockIdx.y * 128;
  const int h  = blockIdx.x;  // 64 columns per head

  v8f acc[4] = {};

  for (int k0 = 0; k0 < D_MODEL; k0 += 64) {
    __syncthreads();
    // Stage A: 128x64 f32 -> bf16 packed, one b64 store per float4
#pragma unroll
    for (int i = 0; i < 8; ++i) {
      const int f = tid + i * 256;
      const int row = f >> 4, c4 = (f & 15) * 4;
      const float4 v = *(const float4*)(X + (size_t)(m0 + row) * D_MODEL + k0 + c4);
      v4bf p; p.x = (__bf16)v.x; p.y = (__bf16)v.y; p.z = (__bf16)v.z; p.w = (__bf16)v.w;
      *(v4bf*)&As[row][c4] = p;
    }
    // Stage B transposed: two consecutive k-rows per thread -> packed pair
    // stores Bt[n][kp] = {W[kp][n], W[kp+1][n]} (b32 each, cvt_pk friendly)
    {
      const int kp = (tid >> 3) * 2, nb = (tid & 7) * 8;
      const float* w0 = W + (size_t)(k0 + kp) * D_MODEL + n0 + nb;
      const float* w1 = w0 + D_MODEL;
      const float4 a0 = *(const float4*)(w0);
      const float4 a1 = *(const float4*)(w0 + 4);
      const float4 b0 = *(const float4*)(w1);
      const float4 b1 = *(const float4*)(w1 + 4);
      *(v2bf*)&Bt[nb + 0][kp] = pack2(a0.x, b0.x);
      *(v2bf*)&Bt[nb + 1][kp] = pack2(a0.y, b0.y);
      *(v2bf*)&Bt[nb + 2][kp] = pack2(a0.z, b0.z);
      *(v2bf*)&Bt[nb + 3][kp] = pack2(a0.w, b0.w);
      *(v2bf*)&Bt[nb + 4][kp] = pack2(a1.x, b1.x);
      *(v2bf*)&Bt[nb + 5][kp] = pack2(a1.y, b1.y);
      *(v2bf*)&Bt[nb + 6][kp] = pack2(a1.z, b1.z);
      *(v2bf*)&Bt[nb + 7][kp] = pack2(a1.w, b1.w);
    }
    __syncthreads();

#pragma unroll
    for (int ks = 0; ks < 2; ++ks) {
      FragBF a;
      const int rowA = wave * 16 + lo;
      a.q[0] = *(const uint4*)&As[rowA][ks * 32 + hi * 16];
      a.q[1] = *(const uint4*)&As[rowA][ks * 32 + hi * 16 + 8];
#pragma unroll
      for (int j = 0; j < 4; ++j) {
        FragBF b;
        const int colB = j * 16 + lo;
        b.q[0] = *(const uint4*)&Bt[colB][ks * 32 + hi * 16];
        b.q[1] = *(const uint4*)&Bt[colB][ks * 32 + hi * 16 + 8];
        acc[j] = wmma_bf16(a.v, b.v, acc[j]);
      }
    }
  }

  // Epilogue. RoPE: cos/sin index i = d%32; pairs (j=0,j=2) share i=lo,
  // (j=1,j=3) share i=lo+16; inv_freq(i+16) = 0.01 * inv_freq(i).
  const size_t obase = (size_t)h * L_SEQ;
  if (apply_rope) {
    const float invf0 = exp2f(-(float)lo * (13.28771238f / 32.0f));
    const float invf1 = invf0 * 0.01f;
#pragma unroll
    for (int r = 0; r < 8; ++r) {
      const int trow = m0 + wave * 16 + r + 8 * hi;
      float sn0, cs0, sn1, cs1;
      __sincosf((float)trow * invf0, &sn0, &cs0);
      __sincosf((float)trow * invf1, &sn1, &cs1);
      const float x0 = acc[0][r], x1 = acc[1][r], x2 = acc[2][r], x3 = acc[3][r];
      __bf16* op = Out + (obase + trow) * D_HEAD + lo;
      op[0]  = (__bf16)(x0 * cs0 - x2 * sn0);
      op[16] = (__bf16)(x1 * cs1 - x3 * sn1);
      op[32] = (__bf16)(x2 * cs0 + x0 * sn0);
      op[48] = (__bf16)(x3 * cs1 + x1 * sn1);
    }
  } else {
#pragma unroll
    for (int r = 0; r < 8; ++r) {
      const int trow = m0 + wave * 16 + r + 8 * hi;
      __bf16* op = Out + (obase + trow) * D_HEAD + lo;
#pragma unroll
      for (int j = 0; j < 4; ++j) op[j * 16] = (__bf16)acc[j][r];
    }
  }
}

// ---------------------------------------------------------------------------
// Kernel 2: flash attention. Block = (head, 128 q rows), 8 waves x 16 rows.
// K tile staged via async global->LDS b128; V transposed with packed pair
// stores; online softmax; causal clip for heads < 8. AO[t][h*64+d] bf16.
// ---------------------------------------------------------------------------
__global__ __launch_bounds__(256) void flash_attn(
    const __bf16* __restrict__ Q, const __bf16* __restrict__ K,
    const __bf16* __restrict__ V, __bf16* __restrict__ AO) {
  __shared__ __bf16 Kt[32][72];      // [key][d]   (B for S = Q K^T)
  __shared__ __bf16 Vt[64][40];      // [d][key]   (B for O = P V)
  __shared__ __bf16 Pb[8][16][40];   // per-wave P staging (C-layout -> A-layout)
  const int tid  = threadIdx.x;
  const int wave = tid >> 5, lane = tid & 31, lo = lane & 15, hi = lane >> 4;
  const int h = blockIdx.y;
  const bool causal = (h < N_AR);
  const int q0 = blockIdx.x * 128;
  const __bf16* Qh = Q + (size_t)h * L_SEQ * D_HEAD;
  const __bf16* Kh = K + (size_t)h * L_SEQ * D_HEAD;
  const __bf16* Vh = V + (size_t)h * L_SEQ * D_HEAD;

  // Q fragments (registers for whole kernel): two 16x32 A-frags
  const int qrow = q0 + wave * 16 + lo;
  FragBF aq[2];
#pragma unroll
  for (int ks = 0; ks < 2; ++ks) {
    aq[ks].q[0] = *(const uint4*)(Qh + (size_t)qrow * D_HEAD + ks * 32 + hi * 16);
    aq[ks].q[1] = *(const uint4*)(Qh + (size_t)qrow * D_HEAD + ks * 32 + hi * 16 + 8);
  }

  v8f o[4] = {};
  float ms[8], ls[8];
#pragma unroll
  for (int r = 0; r < 8; ++r) { ms[r] = -3.0e38f; ls[r] = 0.0f; }

  const int kmax = causal ? (q0 + 128) : L_SEQ;
  for (int kt = 0; kt < kmax; kt += 32) {
    __syncthreads();
    {
      // K tile: pure bf16 copy -> async DMA into LDS (ASYNCcnt path)
      const int kk = tid >> 3, db = (tid & 7) * 8;
      async_copy_b128((unsigned)(uintptr_t)&Kt[kk][db],
                      Kh + (size_t)(kt + kk) * D_HEAD + db);
      // V tile: transpose, two key rows per thread -> packed pair stores
      const int kp = (tid >> 4) * 2, dv = (tid & 15) * 4;
      U2BF u0, u1;
      u0.q = *(const uint2*)(Vh + (size_t)(kt + kp) * D_HEAD + dv);
      u1.q = *(const uint2*)(Vh + (size_t)(kt + kp + 1) * D_HEAD + dv);
#pragma unroll
      for (int j = 0; j < 4; ++j) {
        v2bf p; p.x = u0.h[j]; p.y = u1.h[j];
        *(v2bf*)&Vt[dv + j][kp] = p;
      }
      // Prefetch next tile (global_prefetch_b8; speculative, OOB-safe)
      __builtin_prefetch(Kh + (size_t)(kt + 32 + kk) * D_HEAD + db, 0, 1);
      __builtin_prefetch(Vh + (size_t)(kt + 32 + kp) * D_HEAD + dv, 0, 1);
    }
    wait_async0();
    __syncthreads();

    // S = Q K^T : two 16x16 tiles (keys kt..+15, kt+16..+31)
    v8f s0 = {}, s1 = {};
#pragma unroll
    for (int ks = 0; ks < 2; ++ks) {
      FragBF b0, b1;
      b0.q[0] = *(const uint4*)&Kt[lo][ks * 32 + hi * 16];
      b0.q[1] = *(const uint4*)&Kt[lo][ks * 32 + hi * 16 + 8];
      b1.q[0] = *(const uint4*)&Kt[16 + lo][ks * 32 + hi * 16];
      b1.q[1] = *(const uint4*)&Kt[16 + lo][ks * 32 + hi * 16 + 8];
      s0 = wmma_bf16(aq[ks].v, b0.v, s0);
      s1 = wmma_bf16(aq[ks].v, b1.v, s1);
    }

    // Online softmax update
    const int key0 = kt + lo, key1 = kt + 16 + lo;
#pragma unroll
    for (int r = 0; r < 8; ++r) {
      const int qr = q0 + wave * 16 + r + 8 * hi;
      const bool m0b = causal && (key0 > qr);
      const bool m1b = causal && (key1 > qr);
      float e0 = m0b ? -3.0e38f : s0[r] * 0.125f;
      float e1 = m1b ? -3.0e38f : s1[r] * 0.125f;
      const float mx   = red_max16(fmaxf(e0, e1));
      const float newm = fmaxf(ms[r], mx);
      const float alpha = __expf(ms[r] - newm);
      const float p0 = m0b ? 0.0f : __expf(e0 - newm);
      const float p1 = m1b ? 0.0f : __expf(e1 - newm);
      const float rs = red_sum16(p0 + p1);
      ms[r] = newm;
      ls[r] = ls[r] * alpha + rs;
#pragma unroll
      for (int j = 0; j < 4; ++j) o[j][r] *= alpha;
      Pb[wave][r + 8 * hi][lo]      = (__bf16)p0;
      Pb[wave][r + 8 * hi][16 + lo] = (__bf16)p1;
    }
    asm volatile("s_wait_dscnt 0" ::: "memory");  // intra-wave cross-lane P visibility

    // O += P V : P as A-frag from per-wave LDS, V as B-frags
    FragBF pa;
    pa.q[0] = *(const uint4*)&Pb[wave][lo][hi * 16];
    pa.q[1] = *(const uint4*)&Pb[wave][lo][hi * 16 + 8];
#pragma unroll
    for (int j = 0; j < 4; ++j) {
      FragBF bv;
      bv.q[0] = *(const uint4*)&Vt[j * 16 + lo][hi * 16];
      bv.q[1] = *(const uint4*)&Vt[j * 16 + lo][hi * 16 + 8];
      o[j] = wmma_bf16(pa.v, bv.v, o[j]);
    }
  }

  // Finalize: divide by l, store bf16 to AO[t][h*64 + d]
#pragma unroll
  for (int r = 0; r < 8; ++r) {
    const float inv = 1.0f / ls[r];
    const int trow = q0 + wave * 16 + r + 8 * hi;
    __bf16* op = AO + (size_t)trow * D_MODEL + h * D_HEAD + lo;
#pragma unroll
    for (int j = 0; j < 4; ++j) op[j * 16] = (__bf16)(o[j][r] * inv);
  }
}

// ---------------------------------------------------------------------------
// Kernel 3: out = AO(bf16) @ Wo(f32->bf16), f32 result.
// A tile staged via async global->LDS; K staged 64 per barrier.
// ---------------------------------------------------------------------------
__global__ __launch_bounds__(256) void gemm_out(
    const __bf16* __restrict__ A, const float* __restrict__ W,
    float* __restrict__ Out) {
  __shared__ __bf16 As[128][72];
  __shared__ __bf16 Bt[64][72];
  const int tid  = threadIdx.x;
  const int wave = tid >> 5, lane = tid & 31, lo = lane & 15, hi = lane >> 4;
  const int n0 = blockIdx.x * 64;
  const int m0 = blockIdx.y * 128;

  v8f acc[4] = {};

  for (int k0 = 0; k0 < D_MODEL; k0 += 64) {
    __syncthreads();
    // Stage A (already bf16): 128x64 -> async DMA, 4 x b128 per thread
#pragma unroll
    for (int i = 0; i < 4; ++i) {
      const int f = tid + i * 256;
      const int row = f >> 3, c = (f & 7) * 8;
      async_copy_b128((unsigned)(uintptr_t)&As[row][c],
                      A + (size_t)(m0 + row) * D_MODEL + k0 + c);
    }
    // Stage B transposed with packed-pair f32->bf16 conversion
    {
      const int kp = (tid >> 3) * 2, nb = (tid & 7) * 8;
      const float* w0 = W + (size_t)(k0 + kp) * D_MODEL + n0 + nb;
      const float* w1 = w0 + D_MODEL;
      const float4 a0 = *(const float4*)(w0);
      const float4 a1 = *(const float4*)(w0 + 4);
      const float4 b0 = *(const float4*)(w1);
      const float4 b1 = *(const float4*)(w1 + 4);
      *(v2bf*)&Bt[nb + 0][kp] = pack2(a0.x, b0.x);
      *(v2bf*)&Bt[nb + 1][kp] = pack2(a0.y, b0.y);
      *(v2bf*)&Bt[nb + 2][kp] = pack2(a0.z, b0.z);
      *(v2bf*)&Bt[nb + 3][kp] = pack2(a0.w, b0.w);
      *(v2bf*)&Bt[nb + 4][kp] = pack2(a1.x, b1.x);
      *(v2bf*)&Bt[nb + 5][kp] = pack2(a1.y, b1.y);
      *(v2bf*)&Bt[nb + 6][kp] = pack2(a1.z, b1.z);
      *(v2bf*)&Bt[nb + 7][kp] = pack2(a1.w, b1.w);
    }
    wait_async0();
    __syncthreads();

#pragma unroll
    for (int ks = 0; ks < 2; ++ks) {
      FragBF a;
      const int rowA = wave * 16 + lo;
      a.q[0] = *(const uint4*)&As[rowA][ks * 32 + hi * 16];
      a.q[1] = *(const uint4*)&As[rowA][ks * 32 + hi * 16 + 8];
#pragma unroll
      for (int j = 0; j < 4; ++j) {
        FragBF b;
        const int colB = j * 16 + lo;
        b.q[0] = *(const uint4*)&Bt[colB][ks * 32 + hi * 16];
        b.q[1] = *(const uint4*)&Bt[colB][ks * 32 + hi * 16 + 8];
        acc[j] = wmma_bf16(a.v, b.v, acc[j]);
      }
    }
  }

#pragma unroll
  for (int j = 0; j < 4; ++j) {
#pragma unroll
    for (int r = 0; r < 8; ++r) {
      const int trow = m0 + wave * 16 + r + 8 * hi;
      Out[(size_t)trow * D_MODEL + n0 + j * 16 + lo] = acc[j][r];
    }
  }
}

// ---------------------------------------------------------------------------
extern "C" void kernel_launch(void* const* d_in, const int* in_sizes, int n_in,
                              void* d_out, int out_size, void* d_ws, size_t ws_size,
                              hipStream_t stream) {
  (void)in_sizes; (void)n_in; (void)out_size; (void)ws_size;
  const float* X  = (const float*)d_in[0];
  const float* Wq = (const float*)d_in[1];
  const float* Wk = (const float*)d_in[2];
  const float* Wv = (const float*)d_in[3];
  const float* Wo = (const float*)d_in[4];

  char* ws = (char*)d_ws;
  const size_t hbytes = (size_t)N_HEADS * L_SEQ * D_HEAD * sizeof(__bf16);  // 8 MB
  __bf16* Qb = (__bf16*)(ws);
  __bf16* Kb = (__bf16*)(ws + hbytes);
  __bf16* Vb = (__bf16*)(ws + 2 * hbytes);
  __bf16* AO = (__bf16*)(ws + 3 * hbytes);

  dim3 blk(256);
  dim3 gg(D_MODEL / 64, L_SEQ / 128);  // (16, 32)

  gemm_qkv_rope<<<gg, blk, 0, stream>>>(X, Wq, Qb, 1);
  gemm_qkv_rope<<<gg, blk, 0, stream>>>(X, Wk, Kb, 1);
  gemm_qkv_rope<<<gg, blk, 0, stream>>>(X, Wv, Vb, 0);
  flash_attn<<<dim3(L_SEQ / 128, N_HEADS), blk, 0, stream>>>(Qb, Kb, Vb, AO);
  gemm_out<<<gg, blk, 0, stream>>>(AO, Wo, (float*)d_out);
}
